// HGRNBitAttention_30897994727482
// MI455X (gfx1250) — compile-verified, matrix operand order checked by reading, not run on hardware
//
#include <hip/hip_runtime.h>
#include <cstdint>

// ---------------- problem constants ----------------
#define BATCH   2
#define SEQ     4096
#define DDIM    2048
#define MTOT    (BATCH * SEQ)            // 8192
#define DD      ((size_t)DDIM * DDIM)    // 4194304
#define MD      ((size_t)MTOT * DDIM)    // 16777216
#define NCHUNK  32
#define TCHUNK  (SEQ / NCHUNK)           // 128

typedef int v8i __attribute__((ext_vector_type(8)));
typedef int v4i __attribute__((ext_vector_type(4)));

#define AS1 __attribute__((address_space(1)))
#define AS3 __attribute__((address_space(3)))

#if defined(__has_builtin)
#if __has_builtin(__builtin_amdgcn_global_load_async_to_lds_b128) && \
    __has_builtin(__builtin_amdgcn_s_wait_asynccnt)
#define USE_ASYNC_LDS 1
#else
#define USE_ASYNC_LDS 0
#endif
#else
#define USE_ASYNC_LDS 0
#endif

// ---------------- block reductions (256 threads) ----------------
__device__ __forceinline__ float block_sum256(float v, float* sm) {
    int t = threadIdx.x;
    sm[t] = v; __syncthreads();
    #pragma unroll
    for (int s = 128; s >= 1; s >>= 1) {
        if (t < s) sm[t] += sm[t + s];
        __syncthreads();
    }
    float r = sm[0]; __syncthreads();
    return r;
}

__device__ __forceinline__ float block_max256(float v, float* sm) {
    int t = threadIdx.x;
    sm[t] = v; __syncthreads();
    #pragma unroll
    for (int s = 128; s >= 1; s >>= 1) {
        if (t < s) sm[t] = fmaxf(sm[t], sm[t + s]);
        __syncthreads();
    }
    float r = sm[0]; __syncthreads();
    return r;
}

__device__ __forceinline__ float sigmoidf_(float x) { return 1.0f / (1.0f + expf(-x)); }

// ---------------- weight quantization ----------------
__global__ void zero_ws_kernel(float* p) {
    if (threadIdx.x < 4) p[threadIdx.x] = 0.0f;
}

__global__ __launch_bounds__(256) void wabs_kernel(const float* __restrict__ W, float* __restrict__ dst) {
    __shared__ float sm[256];
    size_t base = (size_t)blockIdx.x * 4096 + threadIdx.x;
    float s = 0.0f;
    #pragma unroll
    for (int i = 0; i < 16; ++i) s += fabsf(W[base + (size_t)i * 256]);
    s = block_sum256(s, sm);
    if (threadIdx.x == 0) atomicAdd(dst, s);
}

__global__ __launch_bounds__(256) void wquant_kernel(const float* __restrict__ W,
                                                     const float* __restrict__ wsumElem,
                                                     float* __restrict__ wdqElem,
                                                     int8_t* __restrict__ q) {
    float mean = *wsumElem * (1.0f / (float)DD);
    float mclip = fmaxf(mean, 1e-5f);
    float s = 1.0f / mclip;
    size_t i0 = ((size_t)blockIdx.x * 256 + threadIdx.x) * 4;
    #pragma unroll
    for (int k = 0; k < 4; ++k) {
        float qq = rintf(W[i0 + k] * s);
        qq = fminf(fmaxf(qq, -1.0f), 1.0f);
        q[i0 + k] = (int8_t)qq;
    }
    if (blockIdx.x == 0 && threadIdx.x == 0) *wdqElem = mclip;
}

// ---------------- activation rmsnorm + int8 absmax quant (one block per token) ----------------
__global__ __launch_bounds__(256) void act_quant_kernel(const float* __restrict__ X,
                                                        const float* __restrict__ nw,
                                                        int8_t* __restrict__ q,
                                                        float* __restrict__ dq) {
    __shared__ float sm[256];
    const int t = threadIdx.x;
    const size_t row = (size_t)blockIdx.x * DDIM;
    float v[8];
    float ss = 0.0f;
    #pragma unroll
    for (int i = 0; i < 8; ++i) { v[i] = X[row + t + i * 256]; ss += v[i] * v[i]; }
    ss = block_sum256(ss, sm);
    float r = rsqrtf(ss * (1.0f / DDIM) + 1e-8f);
    float xn[8]; float amax = 0.0f;
    #pragma unroll
    for (int i = 0; i < 8; ++i) {
        xn[i] = v[i] * r * nw[t + i * 256];
        amax = fmaxf(amax, fabsf(xn[i]));
    }
    amax = block_max256(amax, sm);
    float a = fmaxf(amax, 1e-5f);
    float s = 127.0f / a;
    #pragma unroll
    for (int i = 0; i < 8; ++i) {
        float qq = fminf(fmaxf(rintf(xn[i] * s), -128.0f), 127.0f);
        q[row + t + i * 256] = (int8_t)qq;
    }
    if (t == 0) dq[blockIdx.x] = a * (1.0f / 127.0f);
}

// ---------------- int8 WMMA GEMM: out[m,n] = (sum_k qA[m,k]*qW[n,k]) * adq[m] * (*wdq) ----------------
#define LSTR 80   // padded LDS row stride (bytes) for 64 K-bytes per row

__global__ __launch_bounds__(256) void gemm_i8_wmma(const int8_t* __restrict__ qA,
                                                    const float* __restrict__ adq,
                                                    const int8_t* __restrict__ qW,
                                                    const float* __restrict__ wdq,
                                                    float* __restrict__ out,
                                                    int M, int N, int K) {
    __shared__ int8_t lA[2][128 * LSTR];
    __shared__ int8_t lB[2][128 * LSTR];

    const int tid  = threadIdx.x;
    const int lane = tid & 31;
    const int wave = tid >> 5;          // 0..7
    const int wm   = wave >> 2;         // 0..1 -> 64-row slab
    const int wn   = wave & 3;          // 0..3 -> 32-col slab
    const int bm0  = blockIdx.y * 128;
    const int bn0  = blockIdx.x * 128;

    // global->LDS staging: 256 threads, each moves 32 B of a 128x64 byte tile
    const int ldr  = tid >> 1;          // tile row 0..127
    const int lcol = (tid & 1) * 32;    // byte offset within 64-byte row

    v8i acc[4][2] = {};

    const int8_t* gA = qA + (size_t)(bm0 + ldr) * K + lcol;
    const int8_t* gB = qW + (size_t)(bn0 + ldr) * K + lcol;

    const int lrow = lane & 15;
    const int kgrp = (lane >> 4) * 8;   // ISA 8-bit operand layout: upper half-wave holds K+8 groups
    const int T = K / 64;               // K-steps

#if USE_ASYNC_LDS
    // ---- async direct global->LDS staging, double buffered ----
    #define ISSUE_TILE(buf, koff)                                                          \
        do {                                                                               \
            __builtin_amdgcn_global_load_async_to_lds_b128(                                \
                (AS1 v4i*)(gA + (koff)),                                                   \
                (AS3 v4i*)&lA[(buf)][(size_t)ldr * LSTR + lcol], 0, 0);                    \
            __builtin_amdgcn_global_load_async_to_lds_b128(                                \
                (AS1 v4i*)(gA + (koff) + 16),                                              \
                (AS3 v4i*)&lA[(buf)][(size_t)ldr * LSTR + lcol + 16], 0, 0);               \
            __builtin_amdgcn_global_load_async_to_lds_b128(                                \
                (AS1 v4i*)(gB + (koff)),                                                   \
                (AS3 v4i*)&lB[(buf)][(size_t)ldr * LSTR + lcol], 0, 0);                    \
            __builtin_amdgcn_global_load_async_to_lds_b128(                                \
                (AS1 v4i*)(gB + (koff) + 16),                                              \
                (AS3 v4i*)&lB[(buf)][(size_t)ldr * LSTR + lcol + 16], 0, 0);               \
        } while (0)

    ISSUE_TILE(0, 0);
    for (int it = 0; it < T; ++it) {
        const int cur = it & 1;
        if (it + 1 < T) {
            ISSUE_TILE(cur ^ 1, (it + 1) * 64);
            __builtin_amdgcn_s_wait_asynccnt(4);   // in-order: current buffer's 4 ops retired
        } else {
            __builtin_amdgcn_s_wait_asynccnt(0);
        }
        __syncthreads();                           // all waves' current-buffer DMA complete

        v8i af[4], bf[2];
        #pragma unroll
        for (int i = 0; i < 4; ++i) {
            const int8_t* p = &lA[cur][(size_t)(wm * 64 + i * 16 + lrow) * LSTR + kgrp];
            long long* d = (long long*)&af[i];
            d[0] = *(const long long*)(p);
            d[1] = *(const long long*)(p + 16);
            d[2] = *(const long long*)(p + 32);
            d[3] = *(const long long*)(p + 48);
        }
        #pragma unroll
        for (int j = 0; j < 2; ++j) {
            const int8_t* p = &lB[cur][(size_t)(wn * 32 + j * 16 + lrow) * LSTR + kgrp];
            long long* d = (long long*)&bf[j];
            d[0] = *(const long long*)(p);
            d[1] = *(const long long*)(p + 16);
            d[2] = *(const long long*)(p + 32);
            d[3] = *(const long long*)(p + 48);
        }
        #pragma unroll
        for (int i = 0; i < 4; ++i)
            #pragma unroll
            for (int j = 0; j < 2; ++j)
                acc[i][j] = __builtin_amdgcn_wmma_i32_16x16x64_iu8(
                    true, af[i], true, bf[j], acc[i][j], false, false);

        __syncthreads();                           // buffer safe to overwrite next issue
    }
    #undef ISSUE_TILE
#else
    // ---- fallback: register-staged global->LDS with software pipeline ----
    int4 pa0 = ((const int4*)gA)[0], pa1 = ((const int4*)gA)[1];
    int4 pb0 = ((const int4*)gB)[0], pb1 = ((const int4*)gB)[1];

    for (int it = 0; it < T; ++it) {
        __syncthreads();
        *(int4*)(&lA[0][(size_t)ldr * LSTR + lcol])      = pa0;
        *(int4*)(&lA[0][(size_t)ldr * LSTR + lcol + 16]) = pa1;
        *(int4*)(&lB[0][(size_t)ldr * LSTR + lcol])      = pb0;
        *(int4*)(&lB[0][(size_t)ldr * LSTR + lcol + 16]) = pb1;
        __syncthreads();

        if (it + 1 < T) {
            gA += 64; gB += 64;
            pa0 = ((const int4*)gA)[0]; pa1 = ((const int4*)gA)[1];
            pb0 = ((const int4*)gB)[0]; pb1 = ((const int4*)gB)[1];
        }

        v8i af[4], bf[2];
        #pragma unroll
        for (int i = 0; i < 4; ++i) {
            const int8_t* p = &lA[0][(size_t)(wm * 64 + i * 16 + lrow) * LSTR + kgrp];
            long long* d = (long long*)&af[i];
            d[0] = *(const long long*)(p);
            d[1] = *(const long long*)(p + 16);
            d[2] = *(const long long*)(p + 32);
            d[3] = *(const long long*)(p + 48);
        }
        #pragma unroll
        for (int j = 0; j < 2; ++j) {
            const int8_t* p = &lB[0][(size_t)(wn * 32 + j * 16 + lrow) * LSTR + kgrp];
            long long* d = (long long*)&bf[j];
            d[0] = *(const long long*)(p);
            d[1] = *(const long long*)(p + 16);
            d[2] = *(const long long*)(p + 32);
            d[3] = *(const long long*)(p + 48);
        }
        #pragma unroll
        for (int i = 0; i < 4; ++i)
            #pragma unroll
            for (int j = 0; j < 2; ++j)
                acc[i][j] = __builtin_amdgcn_wmma_i32_16x16x64_iu8(
                    true, af[i], true, bf[j], acc[i][j], false, false);
    }
#endif

    const float wsc = *wdq;
    const int mro = (lane >> 4) * 8;  // C layout: lanes 16-31 hold M+8 rows
    const int nc  = lane & 15;
    #pragma unroll
    for (int i = 0; i < 4; ++i) {
        #pragma unroll
        for (int j = 0; j < 2; ++j) {
            int n = bn0 + wn * 32 + j * 16 + nc;
            #pragma unroll
            for (int r = 0; r < 8; ++r) {
                int m = bm0 + wm * 64 + i * 16 + mro + r;
                out[(size_t)m * N + n] = (float)acc[i][j][r] * (adq[m] * wsc);
            }
        }
    }
}

// ---------------- gating: f=sigmoid(f); i=silu(i)*(1-f) ----------------
__global__ __launch_bounds__(256) void gate_kernel(float* __restrict__ ibuf,
                                                   float* __restrict__ fbuf) {
    size_t idx = (size_t)blockIdx.x * 256 + threadIdx.x;
    float fr = fbuf[idx];
    float ir = ibuf[idx];
    float f  = sigmoidf_(fr);
    float si = ir * sigmoidf_(ir);
    fbuf[idx] = f;
    ibuf[idx] = si * (1.0f - f);
}

// ---------------- chunked HGRN scan h_t = f_t*h_{t-1} + i_t ----------------
__global__ __launch_bounds__(256) void scan_phase1(const float* __restrict__ f,
                                                   const float* __restrict__ i_,
                                                   float* __restrict__ cF,
                                                   float* __restrict__ cI) {
    int tid = blockIdx.x * 256 + threadIdx.x;          // over B*NCHUNK*D
    int d = tid % DDIM;
    int c = (tid / DDIM) % NCHUNK;
    int b = tid / (DDIM * NCHUNK);
    size_t base = ((size_t)b * SEQ + (size_t)c * TCHUNK) * DDIM + d;
    float F = 1.0f, I = 0.0f;
    for (int t = 0; t < TCHUNK; ++t) {
        float fv = f[base + (size_t)t * DDIM];
        float iv = i_[base + (size_t)t * DDIM];
        I = fv * I + iv;
        F = fv * F;
    }
    cF[tid] = F;
    cI[tid] = I;
}

__global__ __launch_bounds__(256) void scan_phase2(const float* __restrict__ cF,
                                                   const float* __restrict__ cI,
                                                   float* __restrict__ cC) {
    int tid = blockIdx.x * 256 + threadIdx.x;          // over B*D
    int d = tid % DDIM;
    int b = tid / DDIM;
    float carry = 0.0f;
    for (int c = 0; c < NCHUNK; ++c) {
        size_t idx = ((size_t)b * NCHUNK + c) * DDIM + d;
        cC[idx] = carry;
        carry = cI[idx] + cF[idx] * carry;
    }
}

__global__ __launch_bounds__(256) void scan_phase3(const float* __restrict__ f,
                                                   float* __restrict__ ih,   // in: i, out: h (in place)
                                                   const float* __restrict__ cC) {
    int tid = blockIdx.x * 256 + threadIdx.x;
    int d = tid % DDIM;
    int c = (tid / DDIM) % NCHUNK;
    int b = tid / (DDIM * NCHUNK);
    size_t base = ((size_t)b * SEQ + (size_t)c * TCHUNK) * DDIM + d;
    float h = cC[tid];
    for (int t = 0; t < TCHUNK; ++t) {
        size_t idx = base + (size_t)t * DDIM;
        h = f[idx] * h + ih[idx];
        ih[idx] = h;
    }
}

// ---------------- o = rmsnorm(g, gw, 1e-5) * h * sigmoid(h); then rmsnorm(o, norm_o, 1e-8) + int8 quant ----------------
__global__ __launch_bounds__(256) void o_quant_kernel(const float* __restrict__ g,
                                                      const float* __restrict__ h,
                                                      const float* __restrict__ gw,
                                                      const float* __restrict__ now,
                                                      int8_t* __restrict__ q,
                                                      float* __restrict__ dq) {
    __shared__ float sm[256];
    const int t = threadIdx.x;
    const size_t row = (size_t)blockIdx.x * DDIM;
    float gv[8], hv[8];
    float ssg = 0.0f;
    #pragma unroll
    for (int i = 0; i < 8; ++i) {
        gv[i] = g[row + t + i * 256];
        hv[i] = h[row + t + i * 256];
        ssg += gv[i] * gv[i];
    }
    ssg = block_sum256(ssg, sm);
    float rg = rsqrtf(ssg * (1.0f / DDIM) + 1e-5f);
    float ov[8];
    float sso = 0.0f;
    #pragma unroll
    for (int i = 0; i < 8; ++i) {
        ov[i] = gv[i] * rg * gw[t + i * 256] * hv[i] * sigmoidf_(hv[i]);
        sso += ov[i] * ov[i];
    }
    sso = block_sum256(sso, sm);
    float ro = rsqrtf(sso * (1.0f / DDIM) + 1e-8f);
    float xn[8]; float amax = 0.0f;
    #pragma unroll
    for (int i = 0; i < 8; ++i) {
        xn[i] = ov[i] * ro * now[t + i * 256];
        amax = fmaxf(amax, fabsf(xn[i]));
    }
    amax = block_max256(amax, sm);
    float a = fmaxf(amax, 1e-5f);
    float s = 127.0f / a;
    #pragma unroll
    for (int i = 0; i < 8; ++i) {
        float qq = fminf(fmaxf(rintf(xn[i] * s), -128.0f), 127.0f);
        q[row + t + i * 256] = (int8_t)qq;
    }
    if (t == 0) dq[blockIdx.x] = a * (1.0f / 127.0f);
}

// ---------------- launch ----------------
extern "C" void kernel_launch(void* const* d_in, const int* in_sizes, int n_in,
                              void* d_out, int out_size, void* d_ws, size_t ws_size,
                              hipStream_t stream) {
    const float* x   = (const float*)d_in[0];
    const float* Wmat[4] = { (const float*)d_in[1], (const float*)d_in[2],
                             (const float*)d_in[3], (const float*)d_in[4] };
    const float* nrm[4]  = { (const float*)d_in[5], (const float*)d_in[6],
                             (const float*)d_in[7], (const float*)d_in[8] };
    const float* gnw = (const float*)d_in[9];
    float* out = (float*)d_out;

    // workspace layout
    char* wsb = (char*)d_ws;
    float*  wsum = (float*)wsb;                        // 4 floats
    float*  wdq  = wsum + 4;                           // 4 floats
    int8_t* qW   = (int8_t*)(wsb + 256);               // 4*DD bytes
    int8_t* qA   = qW + 4 * DD;                        // MD bytes (reused per projection)
    float*  adq  = (float*)(qA + MD);                  // 4*MTOT floats
    float*  buf0 = adq + 4 * (size_t)MTOT;             // MD floats: i -> h
    float*  buf1 = buf0 + MD;                          // MD floats: f
    float*  buf2 = buf1 + MD;                          // MD floats: g
    float*  cF   = buf2 + MD;                          // B*NCHUNK*D floats
    float*  cI   = cF + (size_t)BATCH * NCHUNK * DDIM;
    float*  cC   = cI + (size_t)BATCH * NCHUNK * DDIM;

    // 1) ternary weight quantization (mean|W| then quantize), once per weight
    zero_ws_kernel<<<1, 32, 0, stream>>>(wsum);
    for (int w = 0; w < 4; ++w)
        wabs_kernel<<<1024, 256, 0, stream>>>(Wmat[w], wsum + w);
    for (int w = 0; w < 4; ++w)
        wquant_kernel<<<4096, 256, 0, stream>>>(Wmat[w], wsum + w, wdq + w, qW + (size_t)w * DD);

    // 2) three BitLinear projections of hidden_states: i -> buf0, f -> buf1, g -> buf2
    float* projOut[3] = { buf0, buf1, buf2 };
    dim3 ggrid(DDIM / 128, MTOT / 128);                // (16, 64)
    for (int p = 0; p < 3; ++p) {
        act_quant_kernel<<<MTOT, 256, 0, stream>>>(x, nrm[p], qA, adq + (size_t)p * MTOT);
        gemm_i8_wmma<<<ggrid, 256, 0, stream>>>(qA, adq + (size_t)p * MTOT,
                                                qW + (size_t)p * DD, wdq + p,
                                                projOut[p], MTOT, DDIM, DDIM);
    }

    // 3) gating
    gate_kernel<<<(unsigned)(MD / 256), 256, 0, stream>>>(buf0, buf1);

    // 4) HGRN recurrence (chunked scan), h written in place into buf0
    scan_phase1<<<(BATCH * NCHUNK * DDIM) / 256, 256, 0, stream>>>(buf1, buf0, cF, cI);
    scan_phase2<<<(BATCH * DDIM) / 256, 256, 0, stream>>>(cF, cI, cC);
    scan_phase3<<<(BATCH * NCHUNK * DDIM) / 256, 256, 0, stream>>>(buf1, buf0, cC);

    // 5) o = rmsnorm(g)*h*sigmoid(h), fused with BitLinear input quantization
    o_quant_kernel<<<MTOT, 256, 0, stream>>>(buf2, buf0, gnw, nrm[3], qA, adq + 3 * (size_t)MTOT);

    // 6) final BitLinear -> output
    gemm_i8_wmma<<<ggrid, 256, 0, stream>>>(qA, adq + 3 * (size_t)MTOT,
                                            qW + 3 * DD, wdq + 3,
                                            out, MTOT, DDIM, DDIM);
}
